// DiffMambaLayer_25159918419975
// MI455X (gfx1250) — matrix operand order repaired and checked by reading, not compiled, not run on hardware
//
#include <hip/hip_runtime.h>
#include <hip/hip_bf16.h>

// DiffMamba layer for MI455X (gfx1250, wave32).
// GEMMs: v_wmma_f32_16x16x32_f16, f32 accumulate, branch-free fragment loads
// (float4 activations + pre-packed zero-padded f16 weights). Selective scan is
// lane-parallel over (channel, state) with shfl reductions over the 8 states.
// Working set (~100MB) stays resident in the 192MB L2.

#define DIMC   128
#define DSTATE 8
#define DCONV  4
#define DTRANK 8
#define BATCH  2
#define LSEQ   12288   // 48*16*16
#define EPSLN  1e-5f

typedef __attribute__((ext_vector_type(16))) _Float16 v16h;
typedef __attribute__((ext_vector_type(8)))  float    v8f;

__device__ __forceinline__ float silu_f(float x) {
    return x / (1.f + __expf(-x));
}
__device__ __forceinline__ float softplus_f(float x) {
    return (x > 20.f) ? x : log1pf(__expf(x));
}

// ---------------------------------------------------------------------------
// Pack f32 weight slice [N,K] -> zero-padded f16 [Npad, KPAD] row-major.
// ---------------------------------------------------------------------------
__global__ __launch_bounds__(256)
void pack_w16_kernel(const float* __restrict__ W, _Float16* __restrict__ Wo,
                     int N, int K, int KPAD, int total)
{
    int idx = blockIdx.x * 256 + threadIdx.x;
    if (idx >= total) return;
    int n = idx / KPAD, k = idx % KPAD;
    Wo[idx] = (n < N && k < K) ? (_Float16)W[(size_t)n * K + k] : (_Float16)0.f;
}

// ---------------------------------------------------------------------------
// C[M,N] = f(X[M,K] * Wh[N,KPAD]^T), one 16x16 tile per wave, branch-free.
// A-frag (16-bit 16x32): lane=(M=lane&15, khalf=lane>>4),
//   elem e -> K = khalf*16 + e      (four contiguous float4)
// B-frag (32x16): lane=(N=lane&15, khalf), elem e -> K = khalf*16 + e
//   (one contiguous 32B v16h load from packed f16 weights)
// C/D (f32): VGPR r -> M = r + 8*khalf, N = lane&15.
// KTOT=8 (dt proj): g=1 elems are compile-time zero; g=0 masked by khalf==0.
// OP: 0 = none, 1 = softplus(acc + bias[col]).
// Caller guarantees M % 16 == 0 and Npad % 16 == 0 (no store guards).
// ---------------------------------------------------------------------------
template<int KTOT, int OP>
__global__ __launch_bounds__(256)
void wmma_gemm(const float* __restrict__ X, const _Float16* __restrict__ Wh,
               const float* __restrict__ bias, float* __restrict__ C,
               int ldx, int ldw, int ldc, int tiles_n, int ntiles)
{
    int gt = blockIdx.x * 8 + (threadIdx.x >> 5);
    if (gt >= ntiles) return;
    int tm = (gt / tiles_n) << 4;
    int tn = (gt % tiles_n) << 4;
    int lane  = threadIdx.x & 31;
    int l15   = lane & 15;
    int khalf = lane >> 4;
    constexpr int KPAD = (KTOT + 31) & ~31;

    const float*    arow = X  + (size_t)(tm + l15) * ldx;
    const _Float16* brow = Wh + (size_t)(tn + l15) * ldw;

    v8f acc = {};
#pragma unroll
    for (int k0 = 0; k0 < KPAD; k0 += 32) {
        v16h a;
        if constexpr (KTOT % 32 == 0) {
            const float4 f0 = *(const float4*)(arow + k0 + (khalf << 4));
            const float4 f1 = *(const float4*)(arow + k0 + (khalf << 4) + 4);
            const float4 f2 = *(const float4*)(arow + k0 + (khalf << 4) + 8);
            const float4 f3 = *(const float4*)(arow + k0 + (khalf << 4) + 12);
            a[0]=(_Float16)f0.x;  a[1]=(_Float16)f0.y;  a[2]=(_Float16)f0.z;  a[3]=(_Float16)f0.w;
            a[4]=(_Float16)f1.x;  a[5]=(_Float16)f1.y;  a[6]=(_Float16)f1.z;  a[7]=(_Float16)f1.w;
            a[8]=(_Float16)f2.x;  a[9]=(_Float16)f2.y;  a[10]=(_Float16)f2.z; a[11]=(_Float16)f2.w;
            a[12]=(_Float16)f3.x; a[13]=(_Float16)f3.y; a[14]=(_Float16)f3.z; a[15]=(_Float16)f3.w;
        } else {
            // KTOT == 8: valid data only for khalf==0, elements 0..7 (K=0..7).
            // ldx >= 16 so the loads below are always in-bounds; mask by value.
            const float4 t0 = *(const float4*)(arow + (khalf << 3));
            const float4 t1 = *(const float4*)(arow + (khalf << 3) + 4);
            bool ok = (khalf == 0);
            a[0]=(_Float16)(ok ? t0.x : 0.f); a[1]=(_Float16)(ok ? t0.y : 0.f);
            a[2]=(_Float16)(ok ? t0.z : 0.f); a[3]=(_Float16)(ok ? t0.w : 0.f);
            a[4]=(_Float16)(ok ? t1.x : 0.f); a[5]=(_Float16)(ok ? t1.y : 0.f);
            a[6]=(_Float16)(ok ? t1.z : 0.f); a[7]=(_Float16)(ok ? t1.w : 0.f);
#pragma unroll
            for (int e = 8; e < 16; ++e) a[e] = (_Float16)0.f;
        }
        v16h b = *(const v16h*)(brow + k0 + (khalf << 4));
        acc = __builtin_amdgcn_wmma_f32_16x16x32_f16(false, a, false, b,
                                                     (short)0, acc, false, false);
    }
#pragma unroll
    for (int r = 0; r < 8; ++r) {
        int row = tm + r + (khalf << 3);
        float v = acc[r];
        if constexpr (OP == 1) v = softplus_f(v + bias[tn + l15]);
        C[(size_t)row * ldc + (tn + l15)] = v;
    }
}

// ---------------------------------------------------------------------------
// LN over channels of x[B, DIMC, L] -> xnorm[B*L, DIMC]. One wave per row.
// ---------------------------------------------------------------------------
__global__ __launch_bounds__(256)
void ln_in_kernel(const float* __restrict__ x, const float* __restrict__ w,
                  const float* __restrict__ bb, float* __restrict__ out)
{
    int row = blockIdx.x * 8 + (threadIdx.x >> 5);
    if (row >= BATCH * LSEQ) return;
    int lane = threadIdx.x & 31;
    int b = row / LSEQ, l = row % LSEQ;
    const float* base = x + (size_t)b * DIMC * LSEQ + l;
    float v[4], s = 0.f;
#pragma unroll
    for (int i = 0; i < 4; ++i) {
        int c = lane + (i << 5);
        v[i] = base[(size_t)c * LSEQ];
        s += v[i];
    }
#pragma unroll
    for (int m = 16; m; m >>= 1) s += __shfl_xor(s, m, 32);
    float mu = s * (1.f / DIMC);
    float var = 0.f;
#pragma unroll
    for (int i = 0; i < 4; ++i) { float d = v[i] - mu; var += d * d; }
#pragma unroll
    for (int m = 16; m; m >>= 1) var += __shfl_xor(var, m, 32);
    float rs = rsqrtf(var * (1.f / DIMC) + EPSLN);
#pragma unroll
    for (int i = 0; i < 4; ++i) {
        int c = lane + (i << 5);
        out[(size_t)row * DIMC + c] = (v[i] - mu) * rs * w[c] + bb[c];
    }
}

// ---------------------------------------------------------------------------
// Causal depthwise conv (K=4) + SiLU in scan order p for direction dir
// (p -> original l = dir ? L-1-p : p; tap j reads original index l -/+ j).
// ---------------------------------------------------------------------------
__global__ __launch_bounds__(256)
void conv_silu_kernel(const float* __restrict__ xz, const float* __restrict__ cw,
                      const float* __restrict__ cb, float* __restrict__ xc, int dir)
{
    int idx = blockIdx.x * 256 + threadIdx.x;
    if (idx >= BATCH * LSEQ * DIMC) return;
    int c = idx & (DIMC - 1);
    int p = (idx >> 7) % LSEQ;
    int b = (idx >> 7) / LSEQ;
    int l = dir ? (LSEQ - 1 - p) : p;
    float s = cb[c];
#pragma unroll
    for (int j = 0; j < DCONV; ++j) {
        if (p - j >= 0) {
            int lj = dir ? (l + j) : (l - j);
            s += cw[c * DCONV + (DCONV - 1 - j)] *
                 xz[((size_t)b * LSEQ + lj) * (2 * DIMC) + c];
        }
    }
    xc[((size_t)b * LSEQ + p) * DIMC + c] = silu_f(s);
}

// ---------------------------------------------------------------------------
// Selective scan. Lane = (channel d, state n = lane&7); 32 channels / block.
// h = exp(dt*A)*h + dt*B*u ; y = sum_n h*C + u*D ; out = y*silu(z).
// shfl_xor(width 8) reduces over states. dir==0 writes acc, dir==1 adds.
// ---------------------------------------------------------------------------
__global__ __launch_bounds__(256)
void scan_kernel(const float* __restrict__ xc, const float* __restrict__ dtb,
                 const float* __restrict__ proj, const float* __restrict__ xz,
                 const float* __restrict__ Alog, const float* __restrict__ Dp,
                 float* __restrict__ acc, int dir)
{
    int wave = threadIdx.x >> 5;
    int lane = threadIdx.x & 31;
    int gb = blockIdx.x;          // 0 .. BATCH*4-1
    int b  = gb >> 2;
    int d  = ((gb & 3) << 5) + (wave << 2) + (lane >> 3);
    int n  = lane & 7;

    float A  = -__expf(Alog[d * DSTATE + n]);
    float Dv = Dp[d];
    float h  = 0.f;
    const size_t baseBL = (size_t)b * LSEQ;

    for (int p = 0; p < LSEQ; ++p) {
        size_t r = baseBL + p;
        float dtv = dtb[r * DIMC + d];
        float u   = xc [r * DIMC + d];
        float Bv  = proj[r * 32 + DTRANK + n];
        float Cv  = proj[r * 32 + DTRANK + DSTATE + n];
        float dA  = __expf(dtv * A);
        h = __builtin_fmaf(dA, h, dtv * Bv * u);
        float cv = h * Cv;
        cv += __shfl_xor(cv, 1, 8);
        cv += __shfl_xor(cv, 2, 8);
        cv += __shfl_xor(cv, 4, 8);
        if (n == 0) {
            int lo = dir ? (LSEQ - 1 - p) : p;
            float zv = xz[(baseBL + lo) * (2 * DIMC) + DIMC + d];
            float y  = (cv + u * Dv) * silu_f(zv);
            float* dst = acc + (baseBL + lo) * DIMC + d;
            if (dir) *dst += y; else *dst = y;
        }
    }
}

// ---------------------------------------------------------------------------
// Final: lam = sigmoid(sum lambda_q); attn = LN(y1 - lam*y2, subln);
// out = LN(x + attn, norm2), stored transposed as [B, DIMC, L].
// ---------------------------------------------------------------------------
__global__ __launch_bounds__(256)
void final_kernel(const float* __restrict__ x, const float* __restrict__ y1,
                  const float* __restrict__ y2, const float* __restrict__ lamq,
                  const float* __restrict__ slw, const float* __restrict__ slb,
                  const float* __restrict__ n2w, const float* __restrict__ n2b,
                  float* __restrict__ out)
{
    int row = blockIdx.x * 8 + (threadIdx.x >> 5);
    if (row >= BATCH * LSEQ) return;
    int lane = threadIdx.x & 31;
    int b = row / LSEQ, l = row % LSEQ;

    float ls = 0.f;
#pragma unroll
    for (int i = 0; i < 4; ++i) ls += lamq[lane + (i << 5)];
#pragma unroll
    for (int m = 16; m; m >>= 1) ls += __shfl_xor(ls, m, 32);
    float lam = 1.f / (1.f + __expf(-ls));

    float t[4], s = 0.f;
#pragma unroll
    for (int i = 0; i < 4; ++i) {
        size_t o = (size_t)row * DIMC + lane + (i << 5);
        t[i] = y1[o] - lam * y2[o];
        s += t[i];
    }
#pragma unroll
    for (int m = 16; m; m >>= 1) s += __shfl_xor(s, m, 32);
    float mu = s * (1.f / DIMC), var = 0.f;
#pragma unroll
    for (int i = 0; i < 4; ++i) { float d = t[i] - mu; var += d * d; }
#pragma unroll
    for (int m = 16; m; m >>= 1) var += __shfl_xor(var, m, 32);
    float rs = rsqrtf(var * (1.f / DIMC) + EPSLN);

    float rsd[4]; float s2 = 0.f;
#pragma unroll
    for (int i = 0; i < 4; ++i) {
        int c = lane + (i << 5);
        float attn = (t[i] - mu) * rs * slw[c] + slb[c];
        rsd[i] = x[(size_t)b * DIMC * LSEQ + (size_t)c * LSEQ + l] + attn;
        s2 += rsd[i];
    }
#pragma unroll
    for (int m = 16; m; m >>= 1) s2 += __shfl_xor(s2, m, 32);
    float mu2 = s2 * (1.f / DIMC), var2 = 0.f;
#pragma unroll
    for (int i = 0; i < 4; ++i) { float d = rsd[i] - mu2; var2 += d * d; }
#pragma unroll
    for (int m = 16; m; m >>= 1) var2 += __shfl_xor(var2, m, 32);
    float rs2 = rsqrtf(var2 * (1.f / DIMC) + EPSLN);
#pragma unroll
    for (int i = 0; i < 4; ++i) {
        int c = lane + (i << 5);
        out[(size_t)b * DIMC * LSEQ + (size_t)c * LSEQ + l] =
            (rsd[i] - mu2) * rs2 * n2w[c] + n2b[c];
    }
}

// ---------------------------------------------------------------------------
extern "C" void kernel_launch(void* const* d_in, const int* in_sizes, int n_in,
                              void* d_out, int out_size, void* d_ws, size_t ws_size,
                              hipStream_t stream)
{
    const float* x    = (const float*)d_in[0];
    const float* n1w  = (const float*)d_in[1];
    const float* n1b  = (const float*)d_in[2];
    const float* n2w  = (const float*)d_in[3];
    const float* n2b  = (const float*)d_in[4];
    const float* slw  = (const float*)d_in[5];
    const float* slb  = (const float*)d_in[6];
    const float* lamq = (const float*)d_in[7];
    const float* inw  = (const float*)d_in[8];
    const float* cw   = (const float*)d_in[9];
    const float* cb   = (const float*)d_in[10];
    const float* xpw  = (const float*)d_in[11];
    const float* dtw  = (const float*)d_in[12];
    const float* dtbb = (const float*)d_in[13];
    const float* alog = (const float*)d_in[14];
    const float* Dp   = (const float*)d_in[15];
    const float* outw = (const float*)d_in[16];
    float* out = (float*)d_out;

    const size_t BL = (size_t)BATCH * LSEQ;   // 24576 rows
    float* ws    = (float*)d_ws;
    float* xnorm = ws;  ws += BL * DIMC;       // [BL,128]
    float* xz    = ws;  ws += BL * 2 * DIMC;   // [BL,256] (xh | z)
    float* xc    = ws;  ws += BL * DIMC;       // conv+silu (scan order)
    float* proj  = ws;  ws += BL * 32;         // [BL,32]: dtr(8)|B(8)|C(8)|0(8)
    float* dtv   = ws;  ws += BL * DIMC;       // softplus(dt)
    float* accb  = ws;  ws += BL * DIMC;       // fwd+bwd scan output
    float* y1    = ws;  ws += BL * DIMC;
    float* y2    = ws;  ws += BL * DIMC;       // ~104 MB so far
    _Float16* wh    = (_Float16*)ws;
    _Float16* w_in  = wh;               wh += 2 * 256 * DIMC;   // [512,128]
    _Float16* w_xp  = wh;               wh += 4 * 32 * DIMC;    // 4x [32,128]
    _Float16* w_dt  = wh;               wh += 4 * DIMC * 32;    // 4x [128,32]
    _Float16* w_out = wh;               wh += 2 * DIMC * DIMC;  // [256,128]

    const int rows = (int)BL;
    const int rowBlocks = (rows + 7) / 8;

    auto pack = [&](const float* W, _Float16* Wo, int N, int K, int Npad, int KPAD) {
        int tot = Npad * KPAD;
        pack_w16_kernel<<<(tot + 255) / 256, 256, 0, stream>>>(W, Wo, N, K, KPAD, tot);
    };
    pack(inw,  w_in,  512, 128, 512, 128);            // in_proj, both m
    pack(outw, w_out, 256, 128, 256, 128);            // out_proj, both m
    for (int md = 0; md < 4; ++md) {
        pack(xpw + (size_t)md * 24 * DIMC, w_xp + (size_t)md * 32 * DIMC,
             24, 128, 32, 128);                       // x_proj padded 24->32 rows
        pack(dtw + (size_t)md * DIMC * DTRANK, w_dt + (size_t)md * DIMC * 32,
             128, 8, 128, 32);                        // dt_w padded 8->32 cols
    }

    ln_in_kernel<<<rowBlocks, 256, 0, stream>>>(x, n1w, n1b, xnorm);

    for (int m = 0; m < 2; ++m) {
        {   // in_proj: [BL,128] @ [256,128]^T -> xz [BL,256]
            int tn = 16, nt = (rows / 16) * tn;
            wmma_gemm<128, 0><<<nt / 8, 256, 0, stream>>>(
                xnorm, w_in + (size_t)m * 256 * DIMC, nullptr, xz,
                DIMC, DIMC, 2 * DIMC, tn, nt);
        }
        for (int dir = 0; dir < 2; ++dir) {
            int md = m * 2 + dir;
            conv_silu_kernel<<<(int)((BL * DIMC + 255) / 256), 256, 0, stream>>>(
                xz, cw + (size_t)md * DIMC * DCONV, cb + (size_t)md * DIMC, xc, dir);
            {   // x_proj (N padded to 32): -> proj [BL,32]
                int tn = 2, nt = (rows / 16) * tn;
                wmma_gemm<128, 0><<<nt / 8, 256, 0, stream>>>(
                    xc, w_xp + (size_t)md * 32 * DIMC, nullptr, proj,
                    DIMC, DIMC, 32, tn, nt);
            }
            {   // dt: softplus(proj[:, :8] @ dt_w^T + bias) -> dtv [BL,128]
                int tn = 8, nt = (rows / 16) * tn;
                wmma_gemm<8, 1><<<nt / 8, 256, 0, stream>>>(
                    proj, w_dt + (size_t)md * DIMC * 32, dtbb + (size_t)md * DIMC,
                    dtv, 32, 32, DIMC, tn, nt);
            }
            scan_kernel<<<BATCH * (DIMC / 32), 256, 0, stream>>>(
                xc, dtv, proj, xz, alog + (size_t)md * DIMC * DSTATE,
                Dp + (size_t)md * DIMC, accb, dir);
        }
        {   // out_proj: [BL,128] @ [128,128]^T -> y_m
            int tn = 8, nt = (rows / 16) * tn;
            wmma_gemm<128, 0><<<nt / 8, 256, 0, stream>>>(
                accb, w_out + (size_t)m * DIMC * DIMC, nullptr, m ? y2 : y1,
                DIMC, DIMC, DIMC, tn, nt);
        }
    }

    final_kernel<<<rowBlocks, 256, 0, stream>>>(x, y1, y2, lamq, slw, slb,
                                                n2w, n2b, out);
    (void)in_sizes; (void)n_in; (void)out_size; (void)ws_size;
}
